// GeneratorHead_55671366091456
// MI455X (gfx1250) — matrix operand order repaired and checked by reading, not compiled, not run on hardware
//
#include <hip/hip_runtime.h>

// MI455X / gfx1250, wave32. Modulated-conv head via bf16 WMMA implicit GEMM.
// v4: B fragments software-pipelined across taps (ping-pong register buffers):
//     tap t+1's 4 B fragments are ds_loaded before tap t's 8 WMMAs issue, so
//     both buffers are simultaneously live and cannot be coalesced -> WMMAs
//     never wait on same-tap LDS loads.

typedef __attribute__((ext_vector_type(16))) __bf16 bf16x16;
typedef __attribute__((ext_vector_type(8)))  float  f32x8;

#define CC     512
#define HH     64
#define BBATCH 8
#define EPSV   1e-8f
#define SLOPE  0.2f

__device__ __forceinline__ unsigned short f2bf(float f) {
  unsigned int u = __float_as_uint(f);
  u += 0x7FFFu + ((u >> 16) & 1u);           // round-to-nearest-even
  return (unsigned short)(u >> 16);
}

union FragAB {
  uint4   q[2];    // 32 bytes
  bf16x16 v;       // 16 x bf16
};

// ---------------------------------------------------------------- styles
// s[b,i] = (1/sqrt(512)) * dot(w[b,:], A_w[i,:]) + A_b[i]   (for both A1, A2)
__global__ __launch_bounds__(256) void k_styles(
    const float* __restrict__ w,
    const float* __restrict__ A1w, const float* __restrict__ A1b,
    const float* __restrict__ A2w, const float* __restrict__ A2b,
    float* __restrict__ s1, float* __restrict__ s2) {
  int t = blockIdx.x * 256 + threadIdx.x;
  if (t >= 2 * BBATCH * CC) return;
  int sel = t >> 12;                 // 0: style1, 1: style2
  int r   = t & (BBATCH * CC - 1);
  int b   = r >> 9;
  int i   = r & (CC - 1);
  const float* Aw = sel ? A2w : A1w;
  const float* Ab = sel ? A2b : A1b;
  const float* wp = w + b * CC;
  const float* ap = Aw + (size_t)i * CC;
  float acc = 0.f;
#pragma unroll 8
  for (int j = 0; j < CC; ++j) acc += wp[j] * ap[j];
  float val = acc * 0.04419417382415922f + Ab[i];   // 1/sqrt(512)
  (sel ? s2 : s1)[b * CC + i] = val;
}

// ---------------------------------------------------------------- wsq[o,i] = sum_t W^2
__global__ __launch_bounds__(256) void k_wsq(const float* __restrict__ cw,
                                             float* __restrict__ wsq) {
  int t = blockIdx.x * 256 + threadIdx.x;        // o*512 + i
  if (t >= CC * CC) return;
  const float* p = cw + (size_t)t * 9;
  float s = 0.f;
#pragma unroll
  for (int k = 0; k < 9; ++k) { float v = p[k]; s += v * v; }
  wsq[t] = s;
}

// ---------------------------------------------------------------- d[b,o] = rsqrt(sum_i wsq*s^2 + eps)
__global__ __launch_bounds__(256) void k_demod(const float* __restrict__ wsq,
                                               const float* __restrict__ s1,
                                               float* __restrict__ dcoef) {
  int t = blockIdx.x * 256 + threadIdx.x;        // b*512 + o
  if (t >= BBATCH * CC) return;
  int b = t >> 9, o = t & (CC - 1);
  const float* wp = wsq + (size_t)o * CC;
  const float* sp = s1 + b * CC;
  float acc = 0.f;
#pragma unroll 4
  for (int i = 0; i < CC; ++i) { float ss = sp[i]; acc += wp[i] * ss * ss; }
  dcoef[t] = rsqrtf(acc + EPSV);
}

// ---------------------------------------------------------------- weights fp32 -> bf16, tap-major
// wbf[tap][o][i]
__global__ __launch_bounds__(256) void k_wcvt(const float* __restrict__ cw,
                                              unsigned short* __restrict__ wbf) {
  int t = blockIdx.x * 256 + threadIdx.x;
  if (t >= 9 * CC * CC) return;
  int tap = t / (CC * CC);
  int rem = t - tap * (CC * CC);
  int o = rem >> 9, i = rem & (CC - 1);
  wbf[((size_t)tap * CC + o) * CC + i] = f2bf(cw[((size_t)o * CC + i) * 9 + tap]);
}

// ---------------------------------------------------------------- main 3x3 modulated conv
// Block: 256 thr = 8 waves, computes 128 Cout x 128 px (output rows y0, y0+1), one batch b.
// Wave (mw,yw): mw 0..3 -> 32 Cout rows; yw 0..1 -> output row y0+yw (full 64 px).
// Per wave: 2 (M) x 4 (N) WMMA tiles of 16x16, K-chunks of 32 channels, 9 taps.
__global__ __launch_bounds__(256) void k_conv3x3(
    const float* __restrict__ x, const unsigned short* __restrict__ wbf,
    const float* __restrict__ s1, const float* __restrict__ dcoef,
    const float* __restrict__ bias1, float* __restrict__ feat) {
  // modulated input rows y0-1..y0+2 as bf16, pixel-major / channel-minor, halo cols
  __shared__ __align__(32) unsigned short lds_b[4][66][32];   // 16.9 KB

  const int o0   = blockIdx.x * 128;
  const int y0   = blockIdx.y * 2;
  const int b    = blockIdx.z;
  const int lane = threadIdx.x & 31;
  const int wv   = threadIdx.x >> 5;
  const int mw   = wv >> 1;                  // 0..3 : Cout sub-tile
  const int yw   = wv & 1;                   // 0..1 : which output row
  const int l15  = lane & 15;
  const int kb_a = (lane & 16) ? 8 : 0;      // A frag K base (ISA 7.12.2)
  const int kq_b = (lane & 16) ? 8 : 0;      // B frag K base
  const int rb   = (lane & 16) ? 8 : 0;      // C/D M base

  f32x8 acc[2][4];
  {
    f32x8 z = {0.f, 0.f, 0.f, 0.f, 0.f, 0.f, 0.f, 0.f};
#pragma unroll
    for (int mi = 0; mi < 2; ++mi)
#pragma unroll
      for (int ni = 0; ni < 4; ++ni) acc[mi][ni] = z;
  }

  for (int ic0 = 0; ic0 < CC; ic0 += 32) {
    __syncthreads();
    // ---- stage 4 input rows x 32 channels, modulate by s1, convert to bf16
    for (int idx = threadIdx.x; idx < 4 * 32 * 64; idx += 256) {
      int ry  = idx >> 11;          // 0..3  -> input row y0-1+ry
      int r   = idx & 2047;
      int ic  = r >> 6;             // 0..31
      int px  = r & 63;             // 0..63
      int row = y0 - 1 + ry;
      float v = 0.f;
      if ((unsigned)row < 64u)
        v = x[(((size_t)b * CC + ic0 + ic) * HH + row) * HH + px] * s1[b * CC + ic0 + ic];
      lds_b[ry][px + 1][ic] = f2bf(v);
    }
    if (threadIdx.x < 128) {        // halo columns (zero pad)
      int ry = threadIdx.x >> 5;
      int ic = threadIdx.x & 31;
      lds_b[ry][0][ic]  = 0;
      lds_b[ry][65][ic] = 0;
    }
    // prefetch next chunk's input rows into cache (global_prefetch_b8)
    if (ic0 + 32 < CC) {
      int ry  = threadIdx.x >> 6;   // 0..3
      int px  = threadIdx.x & 63;
      int row = y0 - 1 + ry;
      if ((unsigned)row < 64u)
        __builtin_prefetch(&x[(((size_t)b * CC + ic0 + 32) * HH + row) * HH + px], 0, 1);
    }
    __syncthreads();

    // ---- ping-pong B fragment buffers, pipelined one tap ahead
    FragAB bb[2][4];
#pragma unroll
    for (int ni = 0; ni < 4; ++ni) {            // prologue: tap 0 (dy=0,dx=0)
      const uint4* bp = (const uint4*)&lds_b[yw][ni * 16 + l15][kq_b];
      bb[0][ni].q[0] = bp[0];
      bb[0][ni].q[1] = bp[2];
    }

#pragma unroll
    for (int t = 0; t < 9; ++t) {
      const int cur = t & 1;
      // ---- prefetch tap t+1's B fragments into the other buffer
      if (t < 8) {
        const int dyn = (t + 1) / 3, dxn = (t + 1) % 3;
        const int ryn = yw + dyn;
#pragma unroll
        for (int ni = 0; ni < 4; ++ni) {
          const uint4* bp = (const uint4*)&lds_b[ryn][ni * 16 + l15 + dxn][kq_b];
          bb[cur ^ 1][ni].q[0] = bp[0];
          bb[cur ^ 1][ni].q[1] = bp[2];
        }
      }
      // ---- A fragments: weights [tap][o][i], two 16B loads/lane
      FragAB a0, a1;
      {
        const uint4* ap0 = (const uint4*)(wbf +
            ((size_t)(t * CC) + (o0 + mw * 32 + l15)) * CC + ic0 + kb_a);
        a0.q[0] = ap0[0]; a0.q[1] = ap0[2];
        const uint4* ap1 = (const uint4*)(wbf +
            ((size_t)(t * CC) + (o0 + mw * 32 + 16 + l15)) * CC + ic0 + kb_a);
        a1.q[0] = ap1[0]; a1.q[1] = ap1[2];
      }
      // ---- 8 WMMAs on the current (previously loaded) B buffer
#pragma unroll
      for (int ni = 0; ni < 4; ++ni) {
        acc[0][ni] = __builtin_amdgcn_wmma_f32_16x16x32_bf16(false, a0.v, false, bb[cur][ni].v,
                                                             (short)0, acc[0][ni], false, false);
        acc[1][ni] = __builtin_amdgcn_wmma_f32_16x16x32_bf16(false, a1.v, false, bb[cur][ni].v,
                                                             (short)0, acc[1][ni], false, false);
      }
    }
  }

  // ---- epilogue: demod * acc + bias, LeakyReLU(0.2), store fp32
  const int yout = y0 + yw;
#pragma unroll
  for (int mi = 0; mi < 2; ++mi) {
#pragma unroll
    for (int ni = 0; ni < 4; ++ni) {
      int col = ni * 16 + l15;
#pragma unroll
      for (int r = 0; r < 8; ++r) {
        int o = o0 + mw * 32 + mi * 16 + rb + r;
        float v = acc[mi][ni][r] * dcoef[b * CC + o] + bias1[o];
        v = (v >= 0.f) ? v : SLOPE * v;
        feat[(((size_t)b * CC + o) * HH + yout) * HH + col] = v;
      }
    }
  }
}

// ---------------------------------------------------------------- 1x1 to-RGB (bandwidth-trivial)
__global__ __launch_bounds__(256) void k_rgb(const float* __restrict__ feat,
                                             const float* __restrict__ cw2,
                                             const float* __restrict__ s2,
                                             const float* __restrict__ bias2,
                                             float* __restrict__ rgb) {
  int t = blockIdx.x * 256 + threadIdx.x;      // b*4096 + pix
  if (t >= BBATCH * HH * HH) return;
  int b = t >> 12, pix = t & 4095;
  const float* fp = feat + (size_t)b * CC * 4096 + pix;
  const float* sp = s2 + b * CC;
  float a0 = 0.f, a1 = 0.f, a2 = 0.f;
#pragma unroll 4
  for (int o = 0; o < CC; ++o) {
    float m = fp[(size_t)o * 4096] * sp[o];
    a0 += cw2[o] * m;
    a1 += cw2[CC + o] * m;
    a2 += cw2[2 * CC + o] * m;
  }
  size_t base = (size_t)b * 3 * 4096 + pix;
  float v;
  v = a0 + bias2[0]; rgb[base]            = (v >= 0.f) ? v : SLOPE * v;
  v = a1 + bias2[1]; rgb[base + 4096]     = (v >= 0.f) ? v : SLOPE * v;
  v = a2 + bias2[2]; rgb[base + 2 * 4096] = (v >= 0.f) ? v : SLOPE * v;
}

// ----------------------------------------------------------------
extern "C" void kernel_launch(void* const* d_in, const int* in_sizes, int n_in,
                              void* d_out, int out_size, void* d_ws, size_t ws_size,
                              hipStream_t stream) {
  const float* x   = (const float*)d_in[0];   // [8,512,64,64]
  const float* w   = (const float*)d_in[1];   // [8,512]
  const float* c1w = (const float*)d_in[2];   // [512,512,3,3]
  const float* b1  = (const float*)d_in[3];   // [512]
  const float* A1w = (const float*)d_in[4];   // [512,512]
  const float* A1b = (const float*)d_in[5];   // [512]
  const float* c2w = (const float*)d_in[6];   // [3,512,1,1]
  const float* b2  = (const float*)d_in[7];   // [3]
  const float* A2w = (const float*)d_in[8];   // [512,512]
  const float* A2b = (const float*)d_in[9];   // [512]

  float* out  = (float*)d_out;
  float* feat = out;                                   // [8,512,64,64]
  float* rgb  = out + (size_t)BBATCH * CC * HH * HH;   // [8,3,64,64]

  // workspace layout (~5.8 MB)
  float* s1    = (float*)d_ws;                 // 4096
  float* s2    = s1 + BBATCH * CC;             // 4096
  float* dcoef = s2 + BBATCH * CC;             // 4096
  float* wsq   = dcoef + BBATCH * CC;          // 262144
  unsigned short* wbf = (unsigned short*)(wsq + CC * CC);  // 9*512*512 bf16

  k_styles<<<(2 * BBATCH * CC + 255) / 256, 256, 0, stream>>>(w, A1w, A1b, A2w, A2b, s1, s2);
  k_wsq  <<<(CC * CC) / 256, 256, 0, stream>>>(c1w, wsq);
  k_demod<<<(BBATCH * CC) / 256, 256, 0, stream>>>(wsq, s1, dcoef);
  k_wcvt <<<(9 * CC * CC + 255) / 256, 256, 0, stream>>>(c1w, wbf);

  dim3 grid(CC / 128, HH / 2, BBATCH);         // 4 x 32 x 8
  k_conv3x3<<<grid, 256, 0, stream>>>(x, wbf, s1, dcoef, b1, feat);

  k_rgb<<<(BBATCH * HH * HH) / 256, 256, 0, stream>>>(feat, c2w, s2, b2, rgb);
}
